// Encoder_71949292142781
// MI455X (gfx1250) — matrix operand order repaired and checked by reading, not compiled, not run on hardware
//
#include <hip/hip_runtime.h>
#include <math.h>

#define NN 50000
#define NE 800000
#define LN_EPS 1e-5f

typedef __attribute__((ext_vector_type(16))) _Float16 v16h;
typedef __attribute__((ext_vector_type(8)))  float    v8f;

__device__ __forceinline__ float fast_sigmoid(float x) {
  // v_exp_f32 + v_rcp_f32 (TRANS ops) instead of IEEE div expansion
  return __builtin_amdgcn_rcpf(1.f + __expf(-x));
}

// ---------------------------------------------------------------------------
// WMMA fragment index helpers (CDNA5 16-bit layouts, CDNA5 ISA)
// A 16x32: lane<16 -> K {0..7,16..23}, lane>=16 -> K {8..15,24..31}
// B 32x16: lane<16 -> K {0..7,16..23}, lane>=16 -> K {8..15,24..31} (N = lane%16)
// C 16x16 f32: VGPR j: M = j + (lane<16?0:8), N = lane%16 (per n-tile)
// ---------------------------------------------------------------------------
__device__ __forceinline__ int b_krel(int lane, int e) {
  return ((lane >> 4) << 3) + (e & 7) + ((e >> 3) << 4);
}

// Per-wave GEMM: D[16x64] += A[16 x kchunks*32] * B, A in LDS row-major f16,
// B pre-swizzled fragment-major f16 in global memory.
__device__ __forceinline__ void wave_gemm(const _Float16* __restrict__ sA, int lda,
                                          int kchunks, const _Float16* __restrict__ Bs,
                                          v8f* acc, int lane) {
  const int m   = lane & 15;
  const int kh2 = (lane & 16) ? 4 : 0;  // dword offset of this lane's K-half in A row
  for (int kc = 0; kc < kchunks; ++kc) {
    union { v16h v; unsigned u[8]; } au;
    const unsigned* rowp = (const unsigned*)(sA + m * lda + kc * 32);
#pragma unroll
    for (int j = 0; j < 4; ++j) au.u[j] = rowp[kh2 + j];          // K = khalf+2j,+1
#pragma unroll
    for (int j = 0; j < 4; ++j) au.u[4 + j] = rowp[8 + kh2 + j];  // K = 16+khalf+2j,+1
#pragma unroll
    for (int nt = 0; nt < 4; ++nt) {
      v16h b = *(const v16h*)(Bs + (size_t)((kc * 4 + nt) * 32 + lane) * 16);
      acc[nt] = __builtin_amdgcn_wmma_f32_16x16x32_f16(
          false, au.v, false, b, (short)0, acc[nt], false, false);
    }
  }
}

__device__ __forceinline__ void swish_to_smH(const v8f* acc, const float* __restrict__ b1,
                                             _Float16* __restrict__ sH, int lane) {
  const int rbase = (lane & 16) ? 8 : 0;
  const int cl = lane & 15;
#pragma unroll
  for (int nt = 0; nt < 4; ++nt) {
    float bias = b1[nt * 16 + cl];
#pragma unroll
    for (int j = 0; j < 8; ++j) {
      float x = acc[nt][j] + bias;
      float s = x * fast_sigmoid(x);  // swish
      sH[(rbase + j) * 64 + nt * 16 + cl] = (_Float16)s;
    }
  }
}

__device__ __forceinline__ void acc_to_smO(const v8f* acc, const float* __restrict__ b2,
                                           float* __restrict__ sO, int lane) {
  const int rbase = (lane & 16) ? 8 : 0;
  const int cl = lane & 15;
#pragma unroll
  for (int nt = 0; nt < 4; ++nt) {
    float bias = b2[nt * 16 + cl];
#pragma unroll
    for (int j = 0; j < 8; ++j)
      sO[(rbase + j) * 64 + nt * 16 + cl] = acc[nt][j] + bias;
  }
}

// LayerNorm statistics for one 64-wide row staged in LDS (float4 reads).
__device__ __forceinline__ void ln_stats(const float* __restrict__ pr,
                                         float* mu_out, float* inv_out) {
  const float4* p4 = (const float4*)pr;
  float mu = 0.f, m2 = 0.f;
#pragma unroll
  for (int c = 0; c < 16; ++c) {
    float4 v = p4[c];
    mu += v.x + v.y + v.z + v.w;
    m2 += v.x * v.x + v.y * v.y + v.z * v.z + v.w * v.w;
  }
  mu *= (1.f / 64.f);
  float var = fmaxf(m2 * (1.f / 64.f) - mu * mu, 0.f);
  *mu_out = mu;
  *inv_out = __builtin_amdgcn_rsqf(var + LN_EPS);  // v_rsq_f32
}

// FeedForwardBlock on a 16-row tile: Linear -> swish -> Linear (+bias into sO).
// Block-uniform control flow (contains __syncthreads).
__device__ __forceinline__ void ffb_wave(const _Float16* sA, int lda, int kc1,
                                         _Float16* sH, float* sO,
                                         const _Float16* W1s, const float* b1,
                                         const _Float16* W2s, const float* b2, int lane) {
  const v8f vz = {0.f, 0.f, 0.f, 0.f, 0.f, 0.f, 0.f, 0.f};
  v8f acc[4] = {vz, vz, vz, vz};
  wave_gemm(sA, lda, kc1, W1s, acc, lane);
  swish_to_smH(acc, b1, sH, lane);
  __syncthreads();
  v8f acc2[4] = {vz, vz, vz, vz};
  wave_gemm(sH, 64, 2, W2s, acc2, lane);
  acc_to_smO(acc2, b2, sO, lane);
  __syncthreads();
}

// ---------------------------------------------------------------------------
// Weight prep: f32 row-major [K][64] -> f16 fragment-major, K zero-padded.
// ---------------------------------------------------------------------------
__global__ void wprep_kernel(const float* __restrict__ W, _Float16* __restrict__ dst,
                             int K, int Kpad) {
  int total = Kpad * 64;  // halves
  for (int d = blockIdx.x * blockDim.x + threadIdx.x; d < total;
       d += gridDim.x * blockDim.x) {
    int f = d >> 9;          // fragment = 512 halves (32 lanes x 16)
    int within = d & 511;
    int lane = within >> 4;
    int e = within & 15;
    int kc = f >> 2, nt = f & 3;
    int n = nt * 16 + (lane & 15);
    int k = kc * 32 + b_krel(lane, e);
    dst[d] = (k < K) ? (_Float16)W[k * 64 + n] : (_Float16)0.f;
  }
}

__global__ void zero_kernel(float* __restrict__ p, int n) {
  for (int i = blockIdx.x * blockDim.x + threadIdx.x; i < n;
       i += gridDim.x * blockDim.x)
    p[i] = 0.f;
}

// ---------------------------------------------------------------------------
// Node embedding: nf = FFB_LN(node_features).  Writes f32 nf (3rd output) and
// f16 nf copy to workspace for the edge-gather stage.
// ---------------------------------------------------------------------------
__global__ __launch_bounds__(128) void node_embed_kernel(
    const float* __restrict__ xf, const _Float16* __restrict__ W1s,
    const float* __restrict__ b1, const _Float16* __restrict__ W2s,
    const float* __restrict__ b2, const float* __restrict__ g,
    const float* __restrict__ bt, float* __restrict__ nf_out,
    _Float16* __restrict__ nf16) {
  __shared__ __align__(16) _Float16 sA[4][16 * 32];
  __shared__ __align__(16) _Float16 sH[4][16 * 64];
  __shared__ __align__(16) float sO[4][16 * 64];
  const int wid = threadIdx.x >> 5, lane = threadIdx.x & 31;
  const int w = blockIdx.x * 4 + wid;
  const bool valid = (w < NN / 16);
  const int t = valid ? w : (NN / 16 - 1);
  const int n0 = t * 16;
  // Stage [16 x 16] f32 features -> [16 x 32] f16, K padded with zeros.
  for (int i = lane; i < 16 * 32; i += 32) {
    int r = i >> 5, c = i & 31;
    sA[wid][i] = (c < 16) ? (_Float16)xf[(n0 + r) * 16 + c] : (_Float16)0.f;
  }
  __syncthreads();
  ffb_wave(sA[wid], 32, 1, sH[wid], sO[wid], W1s, b1, W2s, b2, lane);
  if (lane < 16) {
    const float* pr = sO[wid] + lane * 64;
    float mu, inv;
    ln_stats(pr, &mu, &inv);
    if (valid) {
      float* po = nf_out + (size_t)(n0 + lane) * 64;
      _Float16* ph = nf16 + (size_t)(n0 + lane) * 64;
      for (int c = 0; c < 64; ++c) {
        float y = (pr[c] - mu) * inv * g[c] + bt[c];
        po[c] = y;
        ph[c] = (_Float16)y;
      }
    }
  }
}

// ---------------------------------------------------------------------------
// Fused edge pipeline: embed edge FFB+LN -> gather nf16[senders/receivers]
// -> process FFB (K=192)+LN -> write edge_latents + scatter-add into agg.
// ---------------------------------------------------------------------------
__global__ __launch_bounds__(128) void edge_kernel(
    const int* __restrict__ eidx, const float* __restrict__ ef_in,
    const _Float16* __restrict__ nf16, const _Float16* __restrict__ Wee1,
    const float* __restrict__ ee_b1, const _Float16* __restrict__ Wee2,
    const float* __restrict__ ee_b2, const float* __restrict__ ee_g,
    const float* __restrict__ ee_bt, const _Float16* __restrict__ Wpe1,
    const float* __restrict__ pe_b1, const _Float16* __restrict__ Wpe2,
    const float* __restrict__ pe_b2, const float* __restrict__ pe_g,
    const float* __restrict__ pe_bt, float* __restrict__ edge_out,
    float* __restrict__ agg) {
  __shared__ __align__(16) _Float16 sA[4][16 * 192];
  __shared__ __align__(16) _Float16 sH[4][16 * 64];
  __shared__ __align__(16) float sO[4][16 * 64];
  const int wid = threadIdx.x >> 5, lane = threadIdx.x & 31;
  const int t = blockIdx.x * 4 + wid;  // 50000 tiles exactly
  const int e0 = t * 16;
  _Float16* A = sA[wid];
  // Stage raw edge features into cols 0..31 (K=4 padded to 32).
  for (int i = lane; i < 16 * 32; i += 32) {
    int r = i >> 5, c = i & 31;
    A[r * 192 + c] = (c < 4) ? (_Float16)ef_in[(e0 + r) * 4 + c] : (_Float16)0.f;
  }
  __syncthreads();
  // embed_edge_fn
  ffb_wave(A, 192, 1, sH[wid], sO[wid], Wee1, ee_b1, Wee2, ee_b2, lane);
  if (lane < 16) {  // LN -> ef (f16) into edge_in cols 0..63
    const float* pr = sO[wid] + lane * 64;
    float mu, inv;
    ln_stats(pr, &mu, &inv);
    for (int c = 0; c < 64; ++c)
      A[lane * 192 + c] = (_Float16)((pr[c] - mu) * inv * ee_g[c] + ee_bt[c]);
  }
  // Gather nf16[senders] -> cols 64..127, nf16[receivers] -> cols 128..191.
  // nf16 is 6.4 MB -> L2-resident; one dword per lane per row per endpoint.
  for (int r = 0; r < 16; ++r) {
    int s = eidx[e0 + r];
    int d = eidx[NE + e0 + r];
    const unsigned* ps = (const unsigned*)(nf16 + (size_t)s * 64);
    const unsigned* pd = (const unsigned*)(nf16 + (size_t)d * 64);
    unsigned* pa = (unsigned*)(A + r * 192);
    pa[32 + lane] = ps[lane];
    pa[64 + lane] = pd[lane];
  }
  __syncthreads();
  // process_edge_fn (K = 192 -> 6 k-chunks)
  ffb_wave(A, 192, 6, sH[wid], sO[wid], Wpe1, pe_b1, Wpe2, pe_b2, lane);
  if (lane < 16) {  // LN + write + scatter segment-sum (hw f32 atomics in L2)
    const float* pr = sO[wid] + lane * 64;
    float mu, inv;
    ln_stats(pr, &mu, &inv);
    int recv = eidx[NE + e0 + lane];
    float* po = edge_out + (size_t)(e0 + lane) * 64;
    float* pa = agg + (size_t)recv * 64;
    for (int c = 0; c < 64; ++c) {
      float y = (pr[c] - mu) * inv * pe_g[c] + pe_bt[c];
      po[c] = y;
      unsafeAtomicAdd(pa + c, y);
    }
  }
}

// ---------------------------------------------------------------------------
// Node update: FFB_LN([nf, agg]) + nf residual, then output MLP (no LN).
// ---------------------------------------------------------------------------
__global__ __launch_bounds__(128) void node_update_kernel(
    const _Float16* __restrict__ nf16, const float* __restrict__ nf_f32,
    const float* __restrict__ agg, const _Float16* __restrict__ Wpn1,
    const float* __restrict__ pn_b1, const _Float16* __restrict__ Wpn2,
    const float* __restrict__ pn_b2, const float* __restrict__ pn_g,
    const float* __restrict__ pn_bt, const _Float16* __restrict__ Wo1,
    const float* __restrict__ o_b1, const _Float16* __restrict__ Wo2,
    const float* __restrict__ o_b2, float* __restrict__ node_out) {
  __shared__ __align__(16) _Float16 sA[4][16 * 128];
  __shared__ __align__(16) _Float16 sH[4][16 * 64];
  __shared__ __align__(16) float sO[4][16 * 64];
  const int wid = threadIdx.x >> 5, lane = threadIdx.x & 31;
  const int w = blockIdx.x * 4 + wid;
  const bool valid = (w < NN / 16);
  const int t = valid ? w : (NN / 16 - 1);
  const int n0 = t * 16;
  _Float16* A = sA[wid];
  // node_in = [nf (f16 copy), f16(agg)]
  for (int r = 0; r < 16; ++r) {
    const unsigned* ps = (const unsigned*)(nf16 + (size_t)(n0 + r) * 64);
    ((unsigned*)(A + r * 128))[lane] = ps[lane];
  }
  for (int i = lane; i < 16 * 64; i += 32) {
    int r = i >> 6, c = i & 63;
    A[r * 128 + 64 + c] = (_Float16)agg[(size_t)(n0 + r) * 64 + c];
  }
  __syncthreads();
  // process_node_fn (K = 128 -> 4 k-chunks)
  ffb_wave(A, 128, 4, sH[wid], sO[wid], Wpn1, pn_b1, Wpn2, pn_b2, lane);
  if (lane < 16) {  // LN + residual(nf f32) -> f16 into A cols 0..63
    const float* pr = sO[wid] + lane * 64;
    float mu, inv;
    ln_stats(pr, &mu, &inv);
    const float* pn = nf_f32 + (size_t)(n0 + lane) * 64;
    for (int c = 0; c < 64; ++c) {
      float y = (pr[c] - mu) * inv * pn_g[c] + pn_bt[c] + pn[c];
      A[lane * 128 + c] = (_Float16)y;
    }
  }
  __syncthreads();
  // node_output_fn (K = 64, no LayerNorm)
  ffb_wave(A, 128, 2, sH[wid], sO[wid], Wo1, o_b1, Wo2, o_b2, lane);
  if (valid) {
    for (int i = lane; i < 16 * 64; i += 32) {
      int r = i >> 6, c = i & 63;
      node_out[(size_t)(n0 + r) * 64 + c] = sO[wid][r * 64 + c];
    }
  }
}

// ---------------------------------------------------------------------------
extern "C" void kernel_launch(void* const* d_in, const int* in_sizes, int n_in,
                              void* d_out, int out_size, void* d_ws, size_t ws_size,
                              hipStream_t stream) {
  (void)in_sizes; (void)n_in; (void)out_size; (void)ws_size;
  const int* eidx = (const int*)d_in[0];
  const float* efeat = (const float*)d_in[1];
  const float* nfeat = (const float*)d_in[2];
  const float* ee_W1 = (const float*)d_in[3];
  const float* ee_b1 = (const float*)d_in[4];
  const float* ee_W2 = (const float*)d_in[5];
  const float* ee_b2 = (const float*)d_in[6];
  const float* ee_g  = (const float*)d_in[7];
  const float* ee_bt = (const float*)d_in[8];
  const float* en_W1 = (const float*)d_in[9];
  const float* en_b1 = (const float*)d_in[10];
  const float* en_W2 = (const float*)d_in[11];
  const float* en_b2 = (const float*)d_in[12];
  const float* en_g  = (const float*)d_in[13];
  const float* en_bt = (const float*)d_in[14];
  const float* pe_W1 = (const float*)d_in[15];
  const float* pe_b1 = (const float*)d_in[16];
  const float* pe_W2 = (const float*)d_in[17];
  const float* pe_b2 = (const float*)d_in[18];
  const float* pe_g  = (const float*)d_in[19];
  const float* pe_bt = (const float*)d_in[20];
  const float* pn_W1 = (const float*)d_in[21];
  const float* pn_b1 = (const float*)d_in[22];
  const float* pn_W2 = (const float*)d_in[23];
  const float* pn_b2 = (const float*)d_in[24];
  const float* pn_g  = (const float*)d_in[25];
  const float* pn_bt = (const float*)d_in[26];
  const float* out_W1 = (const float*)d_in[27];
  const float* out_b1 = (const float*)d_in[28];
  const float* out_W2 = (const float*)d_in[29];
  const float* out_b2 = (const float*)d_in[30];

  // Workspace layout
  char* ws = (char*)d_ws;
  _Float16* nf16 = (_Float16*)ws;                        // 50000*64*2 = 6.4 MB
  float* agg = (float*)(ws + 6400000);                   // 50000*64*4 = 12.8 MB
  _Float16* wbase = (_Float16*)(ws + 19200000);          // swizzled weights, 96 KB
  _Float16* wee1 = wbase + 0;      // Kpad 32
  _Float16* wee2 = wbase + 2048;   // K 64
  _Float16* wen1 = wbase + 6144;   // Kpad 32
  _Float16* wen2 = wbase + 8192;   // K 64
  _Float16* wpe1 = wbase + 12288;  // K 192
  _Float16* wpe2 = wbase + 24576;  // K 64
  _Float16* wpn1 = wbase + 28672;  // K 128
  _Float16* wpn2 = wbase + 36864;  // K 64
  _Float16* wo1  = wbase + 40960;  // K 64
  _Float16* wo2  = wbase + 45056;  // K 64

  float* out_edge = (float*)d_out;                   // [E, 64]
  float* out_node = out_edge + (size_t)NE * 64;      // [N, 64]
  float* out_nf   = out_node + (size_t)NN * 64;      // [N, 64]

  // Phase 0: weight swizzle + agg zero
  wprep_kernel<<<16, 256, 0, stream>>>(ee_W1, wee1, 4, 32);
  wprep_kernel<<<16, 256, 0, stream>>>(ee_W2, wee2, 64, 64);
  wprep_kernel<<<16, 256, 0, stream>>>(en_W1, wen1, 16, 32);
  wprep_kernel<<<16, 256, 0, stream>>>(en_W2, wen2, 64, 64);
  wprep_kernel<<<48, 256, 0, stream>>>(pe_W1, wpe1, 192, 192);
  wprep_kernel<<<16, 256, 0, stream>>>(pe_W2, wpe2, 64, 64);
  wprep_kernel<<<32, 256, 0, stream>>>(pn_W1, wpn1, 128, 128);
  wprep_kernel<<<16, 256, 0, stream>>>(pn_W2, wpn2, 64, 64);
  wprep_kernel<<<16, 256, 0, stream>>>(out_W1, wo1, 64, 64);
  wprep_kernel<<<16, 256, 0, stream>>>(out_W2, wo2, 64, 64);
  zero_kernel<<<800, 256, 0, stream>>>(agg, NN * 64);

  // Phase 1: node embedding (3125 tiles, 4 waves/block)
  node_embed_kernel<<<(NN / 16 + 3) / 4, 128, 0, stream>>>(
      nfeat, wen1, en_b1, wen2, en_b2, en_g, en_bt, out_nf, nf16);

  // Phase 2: fused edge pipeline (50000 tiles)
  edge_kernel<<<NE / 16 / 4, 128, 0, stream>>>(
      eidx, efeat, nf16, wee1, ee_b1, wee2, ee_b2, ee_g, ee_bt,
      wpe1, pe_b1, wpe2, pe_b2, pe_g, pe_bt, out_edge, agg);

  // Phase 3: node update + residual + output MLP
  node_update_kernel<<<(NN / 16 + 3) / 4, 128, 0, stream>>>(
      nf16, out_nf, agg, wpn1, pn_b1, wpn2, pn_b2, pn_g, pn_bt,
      wo1, out_b1, wo2, out_b2, out_node);
}
